// IMGraphConv_79748952752768
// MI455X (gfx1250) — matrix operand order repaired and checked by reading, not compiled, not run on hardware
//
#include <hip/hip_runtime.h>

// ---------------------------------------------------------------------------
// IMGraphConv for MI455X (gfx1250, wave32, WMMA).
// bf16 WMMA (f32 accumulate) for all large GEMMs; LDS-staged shared A tiles
// for the projections; register-pipelined NT streaming for the 400MB adj GEMM;
// hardware v_sin/v_cos for the Fourier-KAN basis.
// ---------------------------------------------------------------------------

#define N_NODES 10000
#define IN_DIM  2000
#define KPAD    2016      // IN_DIM padded to multiple of 32 (weights zero-padded)
#define NCHUNK  63        // KPAD/32
#define HO      128       // H*OUT
#define HIDN    32
#define GRIDG   200
#define GPADC   7         // 7 chunks of 32 -> 224 padded grid
#define NP      10016     // N padded to multiple of 32 for HinT
#define NBLK    448       // fc repack blocks: 32(i) * 2(d) * 7(chunks)

typedef __bf16 bf16_t;
typedef __attribute__((ext_vector_type(16))) bf16_t v16bf;
typedef __attribute__((ext_vector_type(8)))  bf16_t v8bf;
typedef __attribute__((ext_vector_type(8)))  float  v8f;

__device__ __forceinline__ bf16_t f2bf(float f) { return (bf16_t)f; }  // native cvt, RNE

__device__ __forceinline__ v16bf cvt16(v8f lo, v8f hi) {
  v8bf l = __builtin_convertvector(lo, v8bf);
  v8bf h = __builtin_convertvector(hi, v8bf);
  return __builtin_shufflevector(l, h, 0, 1, 2, 3, 4, 5, 6, 7,
                                 8, 9, 10, 11, 12, 13, 14, 15);
}

__device__ __forceinline__ v8f wmma_bf16(v16bf a, v16bf b, v8f c) {
  return __builtin_amdgcn_wmma_f32_16x16x32_bf16(false, a, false, b, (short)0, c,
                                                 false, false);
}

__device__ __forceinline__ v16bf ldB(const bf16_t* p, int half) {
  v8bf lo = *(const v8bf*)(p + half * 8);
  v8bf hi = *(const v8bf*)(p + 16 + half * 8);
  return __builtin_shufflevector(lo, hi, 0, 1, 2, 3, 4, 5, 6, 7,
                                 8, 9, 10, 11, 12, 13, 14, 15);
}

// ---------------------------------------------------------------------------
// 0) zero the small accumulator region (sumsq_q, sumsq_k, ks_cs, vs_cs, kvs)
// ---------------------------------------------------------------------------
__global__ void zero_accum_kernel(float* acc, int n) {
  int i = blockIdx.x * blockDim.x + threadIdx.x;
  if (i < n) acc[i] = 0.0f;
}

// ---------------------------------------------------------------------------
// 1) weight f32 -> bf16, zero-padded to stride KPAD
// ---------------------------------------------------------------------------
__global__ void cvt_w_kernel(const float* __restrict__ W, bf16_t* __restrict__ Wb,
                             int rows) {
  int idx = blockIdx.x * blockDim.x + threadIdx.x;
  if (idx >= rows * KPAD) return;
  int r = idx / KPAD, k = idx - r * KPAD;
  Wb[idx] = f2bf(k < IN_DIM ? W[(size_t)r * IN_DIM + k] : 0.0f);
}

__global__ void cvt_wout_kernel(const float* __restrict__ W, bf16_t* __restrict__ Wb) {
  int idx = blockIdx.x * blockDim.x + threadIdx.x;
  if (idx < 32 * 32) Wb[idx] = f2bf(W[idx]);
}

// fc[d][j][i][k] -> fcpk blocks: b = (i*2+d)*7 + c ; block = 32 rows (k) x 32 cols (j)
// stored column-major per block so B-operand lane (col j) reads two runs of 8 contiguous k.
__global__ void repack_fc_kernel(const float* __restrict__ fc, bf16_t* __restrict__ fcpk) {
  int idx = blockIdx.x * blockDim.x + threadIdx.x;
  if (idx >= NBLK * 1024) return;
  int b  = idx >> 10;
  int jr = idx & 1023;
  int j  = jr >> 5, r = jr & 31;
  int c  = b % GPADC;
  int id = b / GPADC;
  int d  = id & 1, i = id >> 1;
  int kk = c * 32 + r;
  float v = (kk < GRIDG) ? fc[(((size_t)d * 32 + j) * 32 + i) * GRIDG + kk] : 0.0f;
  fcpk[(size_t)b * 1024 + j * 32 + r] = f2bf(v);
}

// ---------------------------------------------------------------------------
// 2) fused projection GEMM: [Ca|Cb](N, MoA+MoB) = X @ [Wa|Wb]^T + bias
//    One 16-row A tile per block, staged once to LDS as bf16 (double-buffered)
//    and shared by all (MoA+MoB)/16 waves.  BDIM is a template constant so the
//    staging is straight-line code; only the very last chunk is guarded.
// ---------------------------------------------------------------------------
template <int BDIM>
__global__ __launch_bounds__(BDIM)
void proj_fused_kernel(const float* __restrict__ X, const bf16_t* __restrict__ Wcat,
                       const float* __restrict__ biasA, const float* __restrict__ biasB,
                       float* __restrict__ Ca, float* __restrict__ Cb,
                       int MoA, int MoB) {
  __shared__ bf16_t sA[2][512];  // two 16x32 bf16 tiles
  int tid = threadIdx.x;
  int wave = tid >> 5;
  int lane = tid & 31;
  int half = lane >> 4, l16 = lane & 15;
  int m0 = blockIdx.x * 16;
  int n0 = wave * 16;
  const bf16_t* brow = Wcat + (size_t)(n0 + l16) * KPAD;

  // fast staging: all k in-bounds (used for chunks 0..61)
  auto stage_fast = [&](int p, int kc) {
#pragma unroll
    for (int e = tid; e < 512; e += BDIM) {
      int row = e >> 5, kk = e & 31;
      float v = __builtin_nontemporal_load(X + (size_t)(m0 + row) * IN_DIM + kc + kk);
      sA[p][e] = f2bf(v);
    }
  };
  // guarded staging: last chunk (kc = 1984), valid iff (e&31) < 16
  auto stage_tail = [&](int p, int kc) {
#pragma unroll
    for (int e = tid; e < 512; e += BDIM) {
      int row = e >> 5, kk = e & 31;
      float v = X[(size_t)(m0 + row) * IN_DIM + kc + (kk < 16 ? kk : 0)];
      sA[p][e] = f2bf(kk < 16 ? v : 0.0f);
    }
  };

  stage_fast(0, 0);
  __syncthreads();

  v8f acc = {0.f, 0.f, 0.f, 0.f, 0.f, 0.f, 0.f, 0.f};
#pragma unroll 1
  for (int i = 0; i < NCHUNK; ++i) {
    int p = i & 1;
    if (i + 1 < NCHUNK - 1)      stage_fast(1 - p, (i + 1) * 32);
    else if (i + 1 == NCHUNK - 1) stage_tail(1 - p, (i + 1) * 32);
    v8bf lo = *(const v8bf*)(&sA[p][l16 * 32 + half * 8]);
    v8bf hi = *(const v8bf*)(&sA[p][l16 * 32 + 16 + half * 8]);
    v16bf a = __builtin_shufflevector(lo, hi, 0, 1, 2, 3, 4, 5, 6, 7,
                                      8, 9, 10, 11, 12, 13, 14, 15);
    v16bf b = ldB(brow + i * 32, half);
    acc = wmma_bf16(a, b, acc);
    __syncthreads();
  }
  if (n0 < MoA) {
    float bb = biasA[n0 + l16];
#pragma unroll
    for (int r = 0; r < 8; ++r)
      Ca[(size_t)(m0 + half * 8 + r) * MoA + n0 + l16] = acc[r] + bb;
  } else {
    int c0 = n0 - MoA;
    float bb = biasB[c0 + l16];
#pragma unroll
    for (int r = 0; r < 8; ++r)
      Cb[(size_t)(m0 + half * 8 + r) * MoB + c0 + l16] = acc[r] + bb;
  }
}

// ---------------------------------------------------------------------------
// 3) column sums + global sum-of-squares over [N,128]
// ---------------------------------------------------------------------------
__global__ void colsum_sumsq_kernel(const float* __restrict__ X, float* colsum,
                                    float* sumsq, int nRows) {
  __shared__ float red[128];
  int t = threadIdx.x;
  float cs = 0.f, ss = 0.f;
  for (int n = blockIdx.x; n < nRows; n += gridDim.x) {
    float v = X[(size_t)n * HO + t];
    cs += v;
    ss += v * v;
  }
  if (colsum) atomicAdd(&colsum[t], cs);
  if (sumsq) {
    red[t] = ss;
    __syncthreads();
    for (int s = 64; s > 0; s >>= 1) {
      if (t < s) red[t] += red[t + s];
      __syncthreads();
    }
    if (t == 0) atomicAdd(sumsq, red[0]);
  }
}

// ---------------------------------------------------------------------------
// 4) kvs[h,m,d] = sum_n KS[n,h,m] * VS[n,h,d]  (unnormalized; scaled later)
// ---------------------------------------------------------------------------
__global__ void kvs_kernel(const float* __restrict__ KS, const float* __restrict__ VS,
                           float* __restrict__ kvs) {
  __shared__ float sk[128], sv[128];
  int t = threadIdx.x;
  float acc[32];
#pragma unroll
  for (int d = 0; d < 32; ++d) acc[d] = 0.f;
  for (int n = blockIdx.x; n < N_NODES; n += gridDim.x) {
    __syncthreads();
    sk[t] = KS[(size_t)n * HO + t];
    sv[t] = VS[(size_t)n * HO + t];
    __syncthreads();
    float k = sk[t];
    int h = t >> 5;
#pragma unroll
    for (int d = 0; d < 32; ++d) acc[d] += k * sv[h * 32 + d];
  }
#pragma unroll
  for (int d = 0; d < 32; ++d) atomicAdd(&kvs[t * 32 + d], acc[d]);
}

// ---------------------------------------------------------------------------
// 5) Hin[N,32] -> HinT bf16 [32, NP] (zero-padded) : B operand for adj GEMM
// ---------------------------------------------------------------------------
__global__ void transpose_cvt_kernel(const float* __restrict__ Hin, bf16_t* __restrict__ HT) {
  int idx = blockIdx.x * blockDim.x + threadIdx.x;
  if (idx >= 32 * NP) return;
  int j = idx / NP, k = idx - j * NP;
  HT[idx] = f2bf(k < N_NODES ? Hin[(size_t)k * 32 + j] : 0.0f);
}

// ---------------------------------------------------------------------------
// 6) adj GEMM: H2[N,32] = adj[N,N] @ Hin[N,32]   (WMMA bf16, cvt-on-load)
//    Streams 400MB of adj once with NT loads (no reuse; keep L2 for HT/fcpk);
//    register double-buffer hides load latency behind the current WMMA pair.
//    Tail (k 9984..10015): low slice fully valid, high slice fully OOB -> zero,
//    so no per-element predication anywhere.
// ---------------------------------------------------------------------------
__global__ void adj_gemm_kernel(const float* __restrict__ adj, const bf16_t* __restrict__ HT,
                                float* __restrict__ H2) {
  int wave = threadIdx.x >> 5;
  int lane = threadIdx.x & 31;
  int half = lane >> 4, l16 = lane & 15;
  int m0 = (blockIdx.x * 8 + wave) * 16;
  if (m0 >= N_NODES) return;
  const float*  arow = adj + (size_t)(m0 + l16) * N_NODES;
  const bf16_t* b0 = HT + (size_t)l16 * NP;          // cols 0..15
  const bf16_t* b1 = HT + (size_t)(16 + l16) * NP;   // cols 16..31
  v8f c0 = {0.f, 0.f, 0.f, 0.f, 0.f, 0.f, 0.f, 0.f};
  v8f c1 = {0.f, 0.f, 0.f, 0.f, 0.f, 0.f, 0.f, 0.f};
  const int kfull = (N_NODES / 32) * 32;  // 9984

  v8f  alo = __builtin_nontemporal_load((const v8f*)(arow + half * 8));
  v8f  ahi = __builtin_nontemporal_load((const v8f*)(arow + 16 + half * 8));
  v16bf vb0 = ldB(b0, half);
  v16bf vb1 = ldB(b1, half);
#pragma unroll 1
  for (int kc = 32; kc < kfull; kc += 32) {
    v8f  nlo = __builtin_nontemporal_load((const v8f*)(arow + kc + half * 8));
    v8f  nhi = __builtin_nontemporal_load((const v8f*)(arow + kc + 16 + half * 8));
    v16bf nb0 = ldB(b0 + kc, half);
    v16bf nb1 = ldB(b1 + kc, half);
    v16bf a = cvt16(alo, ahi);
    c0 = wmma_bf16(a, vb0, c0);
    c1 = wmma_bf16(a, vb1, c1);
    alo = nlo; ahi = nhi; vb0 = nb0; vb1 = nb1;
  }
  {  // last full chunk
    v16bf a = cvt16(alo, ahi);
    c0 = wmma_bf16(a, vb0, c0);
    c1 = wmma_bf16(a, vb1, c1);
  }
  {  // tail chunk k = 9984..10015: lo slice valid, hi slice OOB -> zeros
    v8f tlo = __builtin_nontemporal_load((const v8f*)(arow + kfull + half * 8));
    v8f z = {0.f, 0.f, 0.f, 0.f, 0.f, 0.f, 0.f, 0.f};
    v16bf a = cvt16(tlo, z);
    v16bf tb0 = ldB(b0 + kfull, half);
    v16bf tb1 = ldB(b1 + kfull, half);
    c0 = wmma_bf16(a, tb0, c0);
    c1 = wmma_bf16(a, tb1, c1);
  }
#pragma unroll
  for (int r = 0; r < 8; ++r) {
    H2[(size_t)(m0 + half * 8 + r) * 32 + l16]      = c0[r];
    H2[(size_t)(m0 + half * 8 + r) * 32 + 16 + l16] = c1[r];
  }
}

// ---------------------------------------------------------------------------
// 7) fused Fourier-KAN: one angle-mul pass feeds both cos and sin A tiles,
//    4 WMMAs per (i,chunk); then LDS transpose + W_out WMMA.
// ---------------------------------------------------------------------------
__global__ void kan_kernel(const float* __restrict__ H2, const bf16_t* __restrict__ fcpk,
                           const float* __restrict__ kan_bias, const bf16_t* __restrict__ WoutB,
                           float* __restrict__ kan_out) {
  constexpr float INV2PI = 0.15915493667125702f;  // v_sin/v_cos take revolutions
  __shared__ float lds[8 * 512];                  // per-wave 16x32 f32 slice
  int wave = threadIdx.x >> 5;
  int lane = threadIdx.x & 31;
  int half = lane >> 4, l16 = lane & 15;
  int m0 = (blockIdx.x * 8 + wave) * 16;
  if (m0 >= N_NODES) return;
  float* myl = lds + wave * 512;
  const float* hrow = H2 + (size_t)(m0 + l16) * 32;
  v8f c0 = {0.f, 0.f, 0.f, 0.f, 0.f, 0.f, 0.f, 0.f};
  v8f c1 = {0.f, 0.f, 0.f, 0.f, 0.f, 0.f, 0.f, 0.f};
#pragma unroll 1
  for (int i = 0; i < 32; ++i) {
    float hv = hrow[i] * INV2PI;
#pragma unroll 1
    for (int c = 0; c < GPADC; ++c) {
      int kb = c * 32;
      v16bf ac, as;
#pragma unroll
      for (int t = 0; t < 8; ++t) {
        float ang0 = hv * (float)(kb + half * 8 + t + 1);
        float ang1 = hv * (float)(kb + 16 + half * 8 + t + 1);
        ac[t]     = f2bf(__builtin_amdgcn_cosf(ang0));
        ac[8 + t] = f2bf(__builtin_amdgcn_cosf(ang1));
        as[t]     = f2bf(__builtin_amdgcn_sinf(ang0));
        as[8 + t] = f2bf(__builtin_amdgcn_sinf(ang1));
      }
      const bf16_t* bpc = fcpk + (size_t)((i * 2 + 0) * GPADC + c) * 1024;
      const bf16_t* bps = fcpk + (size_t)((i * 2 + 1) * GPADC + c) * 1024;
      v16bf bc0 = ldB(bpc + l16 * 32, half);
      v16bf bc1 = ldB(bpc + (16 + l16) * 32, half);
      v16bf bs0 = ldB(bps + l16 * 32, half);
      v16bf bs1 = ldB(bps + (16 + l16) * 32, half);
      c0 = wmma_bf16(ac, bc0, c0);
      c1 = wmma_bf16(ac, bc1, c1);
      c0 = wmma_bf16(as, bs0, c0);
      c1 = wmma_bf16(as, bs1, c1);
    }
  }
  // add kan_bias (per output column)
  float kb0 = kan_bias[l16], kb1 = kan_bias[16 + l16];
#pragma unroll
  for (int r = 0; r < 8; ++r) { c0[r] += kb0; c1[r] += kb1; }
  // C-layout -> row-major 16x32 in this wave's private LDS slice
#pragma unroll
  for (int r = 0; r < 8; ++r) {
    myl[(half * 8 + r) * 32 + l16]      = c0[r];
    myl[(half * 8 + r) * 32 + 16 + l16] = c1[r];
  }
  asm volatile("s_wait_dscnt 0" ::: "memory");  // same-wave LDS RAW fence
  // reload in A-operand layout (lane = node row l16; j split by half)
  v16bf ah;
#pragma unroll
  for (int t = 0; t < 8; ++t) {
    ah[t]     = f2bf(myl[l16 * 32 + half * 8 + t]);
    ah[8 + t] = f2bf(myl[l16 * 32 + 16 + half * 8 + t]);
  }
  v16bf bw0 = ldB(WoutB + l16 * 32, half);
  v16bf bw1 = ldB(WoutB + (16 + l16) * 32, half);
  v8f o0 = {0.f, 0.f, 0.f, 0.f, 0.f, 0.f, 0.f, 0.f};
  v8f o1 = {0.f, 0.f, 0.f, 0.f, 0.f, 0.f, 0.f, 0.f};
  o0 = wmma_bf16(ah, bw0, o0);
  o1 = wmma_bf16(ah, bw1, o1);
#pragma unroll
  for (int r = 0; r < 8; ++r) {
    kan_out[(size_t)(m0 + half * 8 + r) * 32 + l16]      = o0[r];
    kan_out[(size_t)(m0 + half * 8 + r) * 32 + 16 + l16] = o1[r];
  }
}

// ---------------------------------------------------------------------------
// 8) final: attention numerator/denominator, global-norm scaling, head mean,
//    KAN residual.  one node per block, thread t = (h,d)
// ---------------------------------------------------------------------------
__global__ void final_kernel(const float* __restrict__ QS, const float* __restrict__ kvs,
                             const float* __restrict__ ks_cs, const float* __restrict__ vs_cs,
                             const float* __restrict__ ssq, const float* __restrict__ ssk,
                             const float* __restrict__ kan_out, float* __restrict__ out) {
  __shared__ float q[128];
  __shared__ float red[128];
  int n = blockIdx.x;
  int t = threadIdx.x;
  q[t] = QS[(size_t)n * HO + t];
  __syncthreads();
  int h = t >> 5, d = t & 31;
  float invq = rsqrtf(*ssq);
  float invk = rsqrtf(*ssk);
  float s = invq * invk;
  float num = 0.f, den = 0.f;
#pragma unroll
  for (int m = 0; m < 32; ++m) {
    float qm = q[h * 32 + m];
    num += qm * kvs[(h * 32 + m) * 32 + d];
    den += qm * ks_cs[h * 32 + m];
  }
  float attn = (num * s + vs_cs[h * 32 + d]) / (den * s + (float)N_NODES);
  red[t] = attn;
  __syncthreads();
  if (h == 0)
    out[(size_t)n * 32 + d] =
        0.25f * (red[d] + red[32 + d] + red[64 + d] + red[96 + d]) +
        kan_out[(size_t)n * 32 + d];
}

// ---------------------------------------------------------------------------
extern "C" void kernel_launch(void* const* d_in, const int* in_sizes, int n_in,
                              void* d_out, int out_size, void* d_ws, size_t ws_size,
                              hipStream_t stream) {
  (void)in_sizes; (void)n_in; (void)out_size; (void)ws_size;
  const float* Xq   = (const float*)d_in[0];
  const float* Xs   = (const float*)d_in[1];
  const float* adj  = (const float*)d_in[2];
  const float* Wq   = (const float*)d_in[3];
  const float* bq   = (const float*)d_in[4];
  const float* Wk   = (const float*)d_in[5];
  const float* bk   = (const float*)d_in[6];
  const float* Wv   = (const float*)d_in[7];
  const float* bv   = (const float*)d_in[8];
  const float* Win  = (const float*)d_in[9];
  const float* bin  = (const float*)d_in[10];
  const float* fc   = (const float*)d_in[11];
  const float* kbia = (const float*)d_in[12];
  const float* Wout = (const float*)d_in[13];
  float* out = (float*)d_out;

  char* ws = (char*)d_ws;
  size_t off = 0;
  auto take = [&](size_t bytes) -> void* {
    void* p = ws + off;
    off += (bytes + 255) & ~(size_t)255;
    return p;
  };
  float*  accum = (float*)take(4354 * sizeof(float));
  float*  ssq   = accum + 0;
  float*  ssk   = accum + 1;
  float*  ks_cs = accum + 2;
  float*  vs_cs = accum + 130;
  float*  kvs   = accum + 258;
  bf16_t* Wqin  = (bf16_t*)take((size_t)160 * KPAD * 2);  // [Wq(128) | W_in(32)]
  bf16_t* Wkv   = (bf16_t*)take((size_t)256 * KPAD * 2);  // [Wk(128) | Wv(128)]
  bf16_t* WoutB = (bf16_t*)take(32 * 32 * 2);
  bf16_t* fcpk  = (bf16_t*)take((size_t)NBLK * 1024 * 2);
  float*  QS    = (float*)take((size_t)N_NODES * HO * 4);
  float*  KS    = (float*)take((size_t)N_NODES * HO * 4);
  float*  VS    = (float*)take((size_t)N_NODES * HO * 4);
  float*  Hin   = (float*)take((size_t)N_NODES * 32 * 4);
  bf16_t* HT    = (bf16_t*)take((size_t)32 * NP * 2);
  float*  H2    = (float*)take((size_t)N_NODES * 32 * 4);
  float*  KanO  = (float*)take((size_t)N_NODES * 32 * 4);

  // 0) zero accumulators (must re-zero every call)
  zero_accum_kernel<<<(4354 + 255) / 256, 256, 0, stream>>>(accum, 4354);
  // 1) weight conversion / repack (into concatenated buffers)
  cvt_w_kernel<<<(HO * KPAD + 255) / 256, 256, 0, stream>>>(Wq, Wqin, HO);
  cvt_w_kernel<<<(HIDN * KPAD + 255) / 256, 256, 0, stream>>>(Win, Wqin + (size_t)HO * KPAD, HIDN);
  cvt_w_kernel<<<(HO * KPAD + 255) / 256, 256, 0, stream>>>(Wk, Wkv, HO);
  cvt_w_kernel<<<(HO * KPAD + 255) / 256, 256, 0, stream>>>(Wv, Wkv + (size_t)HO * KPAD, HO);
  cvt_wout_kernel<<<4, 256, 0, stream>>>(Wout, WoutB);
  repack_fc_kernel<<<(NBLK * 1024) / 256, 256, 0, stream>>>(fc, fcpk);
  // 2) fused projections (WMMA, LDS-staged shared A): Q+W_in over Xq, K+V over Xs
  proj_fused_kernel<320><<<N_NODES / 16, 320, 0, stream>>>(Xq, Wqin, bq, bin, QS, Hin, HO, HIDN);
  proj_fused_kernel<512><<<N_NODES / 16, 512, 0, stream>>>(Xs, Wkv, bk, bv, KS, VS, HO, HO);
  // 3) reductions
  colsum_sumsq_kernel<<<128, 128, 0, stream>>>(QS, nullptr, ssq, N_NODES);
  colsum_sumsq_kernel<<<128, 128, 0, stream>>>(KS, ks_cs, ssk, N_NODES);
  colsum_sumsq_kernel<<<128, 128, 0, stream>>>(VS, vs_cs, nullptr, N_NODES);
  kvs_kernel<<<40, 128, 0, stream>>>(KS, VS, kvs);
  // 4) adjacency propagation (WMMA, bandwidth-bound on 400MB adj, NT stream)
  transpose_cvt_kernel<<<(32 * NP + 255) / 256, 256, 0, stream>>>(Hin, HT);
  adj_gemm_kernel<<<(N_NODES / 16 + 7) / 8, 256, 0, stream>>>(adj, HT, H2);
  // 5) fused Fourier-KAN (sin/cos + WMMA + LDS transpose + W_out WMMA)
  kan_kernel<<<(N_NODES / 16 + 7) / 8, 256, 0, stream>>>(H2, fcpk, kbia, WoutB, KanO);
  // 6) attention combine + head mean + residual
  final_kernel<<<N_NODES, 128, 0, stream>>>(QS, kvs, ks_cs, vs_cs, ssq, ssk, KanO, out);
}
